// RealHolographicMemory_69741678952961
// MI455X (gfx1250) — compile-verified, matrix operand order checked by reading, not run on hardware
//
#include <hip/hip_runtime.h>

typedef __attribute__((ext_vector_type(16))) _Float16 v16h;
typedef __attribute__((ext_vector_type(8)))  _Float16 v8h;
typedef __attribute__((ext_vector_type(8)))  float    v8f;

#define DIMD   1024
#define BATCH  32768
#define DECAYF 0.98f
#define EPSF   1e-8f
#define LSTR   40        // LDS row stride in halves (80B: 16B-aligned, conflict-friendly)
#define RSTR   2056      // replica row stride in halves (4112B, 16B-aligned rows)
#define KCHUNK 4096      // split-K chunk (8 chunks over B=32768)
#define NSPLIT 8

// Pack two f32 -> two f16 in one dword (for single ds_store_b32 staging stores).
__device__ __forceinline__ unsigned pack2h(float lo, float hi) {
  union { _Float16 h[2]; unsigned u; } p;
  p.h[0] = (_Float16)lo; p.h[1] = (_Float16)hi;
  return p.u;
}

// Build a v16h A/B fragment for v_wmma_f32_16x16x32_f16 from an LDS tile stored
// as [row][k] with stride LSTR halves. Per CDNA5 ISA (16-bit A 16x32):
//   lanes 0-15 : K = 0..7 (elems 0-7), K = 16..23 (elems 8-15)
//   lanes 16-31: K = 8..15,            K = 24..31
// row = lane & 15. 16B alignment holds: (r*LSTR + koff)*2 % 16 == 0.
__device__ __forceinline__ v16h frag_ld(const _Float16* base, int lane) {
  int r    = lane & 15;
  int koff = (lane < 16) ? 0 : 8;
  const _Float16* p = base + r * LSTR + koff;
  v8h lo = *(const v8h*)(p);
  v8h hi = *(const v8h*)(p + 16);
  v16h o;
#pragma unroll
  for (int t = 0; t < 8; ++t) { o[t] = lo[t]; o[t + 8] = hi[t]; }
  return o;
}

__device__ __forceinline__ v16h frag_cat(v8h lo, v8h hi) {
  v16h o;
#pragma unroll
  for (int t = 0; t < 8; ++t) { o[t] = lo[t]; o[t + 8] = hi[t]; }
  return o;
}

// ---------------------------------------------------------------------------
// GEMM1: Gp[kc] tile = (1/B) * S^T[chunk] * K[chunk]   (128x128 per WG)
// grid = (8 tj, 8 ti, 8 kc), 256 threads (8 waves, each 32x64 of the tile)
// ---------------------------------------------------------------------------
__global__ __launch_bounds__(256)
void gemm1_corr(const float* __restrict__ S, const float* __restrict__ K,
                float* __restrict__ Gp) {
  __shared__ _Float16 sT[128 * LSTR];   // [i][k] = S[kbase+k][i0+i], f16
  __shared__ _Float16 kT[128 * LSTR];   // [j][k] = K[kbase+k][j0+j], f16

  const int tid  = threadIdx.x;
  const int lane = tid & 31;
  const int w    = tid >> 5;
  const int wr   = w & 3, wc = w >> 2;
  const int ioff = wr * 32, joff = wc * 64;
  const int j0     = blockIdx.x * 128;
  const int i0     = blockIdx.y * 128;
  const int kbase0 = blockIdx.z * KCHUNK;

  v8f acc[8];
#pragma unroll
  for (int t = 0; t < 8; ++t) { v8f z = {}; acc[t] = z; }

  for (int kb = 0; kb < KCHUNK; kb += 32) {
    __syncthreads();
    // Stage 32(k) x 128 blocks of S and K, f32 -> f16, transposed to [col][k].
    // k-pairs packed into one dword => single aligned ds_store_b32 each.
    // Global reads coalesced along the 128-wide column run.
#pragma unroll
    for (int it = 0; it < 8; ++it) {
      int e  = tid + it * 256;          // pair index: e = kp*128 + c
      int kp = e >> 7, c = e & 127;
      int k  = kp * 2;
      long row0 = (long)(kbase0 + kb + k) * DIMD;
      float s0 = S[row0 + i0 + c], s1 = S[row0 + DIMD + i0 + c];
      float k0 = K[row0 + j0 + c], k1 = K[row0 + DIMD + j0 + c];
      *(unsigned*)&sT[c * LSTR + k] = pack2h(s0, s1);
      *(unsigned*)&kT[c * LSTR + k] = pack2h(k0, k1);
    }
    __syncthreads();

    v16h a0 = frag_ld(&sT[(ioff     ) * LSTR], lane);
    v16h a1 = frag_ld(&sT[(ioff + 16) * LSTR], lane);
#pragma unroll
    for (int bj = 0; bj < 4; ++bj) {
      v16h b = frag_ld(&kT[(joff + bj * 16) * LSTR], lane);
      acc[0 * 4 + bj] = __builtin_amdgcn_wmma_f32_16x16x32_f16(
          false, a0, false, b, (short)0, acc[0 * 4 + bj], false, false);
      acc[1 * 4 + bj] = __builtin_amdgcn_wmma_f32_16x16x32_f16(
          false, a1, false, b, (short)0, acc[1 * 4 + bj], false, false);
    }
  }

  // Each (ti,tj,kc) owns a disjoint region of its partial buffer: plain stores,
  // fully deterministic (no float atomics).
  float* Gout = Gp + (size_t)blockIdx.z * DIMD * DIMD;
  const float scale = 1.0f / (float)BATCH;
  const int r16 = lane >> 4, nloc = lane & 15;
#pragma unroll
  for (int ai = 0; ai < 2; ++ai)
#pragma unroll
    for (int bj = 0; bj < 4; ++bj)
#pragma unroll
      for (int v = 0; v < 8; ++v) {
        int gi = i0 + ioff + ai * 16 + v + 8 * r16;   // C layout: M = v (+8 hi half)
        int gj = j0 + joff + bj * 16 + nloc;          // N = lane & 15
        Gout[(size_t)gi * DIMD + gj] = acc[ai * 4 + bj][v] * scale;
      }
}

// ---------------------------------------------------------------------------
// Reduce the NSPLIT partial G buffers into Gs.
// ---------------------------------------------------------------------------
__global__ __launch_bounds__(256)
void reduceG(const float* __restrict__ Gp, float* __restrict__ Gs) {
  int i = blockIdx.x * 256 + threadIdx.x;
  float s = 0.f;
#pragma unroll
  for (int p = 0; p < NSPLIT; ++p) s += Gp[(size_t)p * DIMD * DIMD + i];
  Gs[i] = s;
}

// ---------------------------------------------------------------------------
// finalize: update[n] = sum_m G[m][(n-m) mod D]; mem = memory*DECAY + update;
// mem /= (mean|mem| + eps); emit duplicated f16 memExt[2048] for the circulant.
// One workgroup of 1024 threads.
// ---------------------------------------------------------------------------
__global__ __launch_bounds__(1024)
void finalize_mem(const float* __restrict__ G, const float* __restrict__ memory,
                  _Float16* __restrict__ memExt) {
  __shared__ float red[1024];
  const int n = threadIdx.x;
  float s = 0.f;
  for (int m = 0; m < DIMD; ++m)
    s += G[(size_t)m * DIMD + ((n - m) & (DIMD - 1))];
  float mem = memory[n] * DECAYF + s;
  red[n] = fabsf(mem);
  __syncthreads();
  for (int st = 512; st > 0; st >>= 1) {
    if (n < st) red[n] += red[n + st];
    __syncthreads();
  }
  float norm = red[0] * (1.0f / DIMD) + EPSF;
  mem = mem / norm;
  memExt[n]        = (_Float16)mem;
  memExt[n + DIMD] = (_Float16)mem;
}

// ---------------------------------------------------------------------------
// GEMM2: out = K (32768x1024) x Circulant(mem); B[k][n] = memExt[k+n].
// The circulant fragment base (n + kb + koff) has k-loop-invariant low bits,
// so 8 shifted LDS replicas of memExt (rep_a[t] = memExt[t+a], a = n&7) turn
// every B-fragment half into one aligned ds_load_b128.
// grid = (8 tj, 256 ti), 256 threads; 128x128 tile per WG.
// ---------------------------------------------------------------------------
__global__ __launch_bounds__(256)
void gemm2_read(const float* __restrict__ K, const _Float16* __restrict__ memExt,
                float* __restrict__ out) {
  __shared__ _Float16 aT[128 * LSTR];   // [i][k] = K[i0+i][kb+k], f16
  __shared__ _Float16 rep[8 * RSTR];    // 8 alignment-shifted replicas of memExt

  const int tid  = threadIdx.x;
  const int lane = tid & 31;
  const int w    = tid >> 5;
  const int wr   = w & 3, wc = w >> 2;
  const int ioff = wr * 32, joff = wc * 64;
  const int n0 = blockIdx.x * 128;
  const int i0 = blockIdx.y * 128;

  // Build the 8 shifted replicas (indices used never exceed memExt[2046]).
  for (int idx = tid; idx < 8 * 2048; idx += 256) {
    int a = idx >> 11, t = idx & 2047;
    rep[a * RSTR + t] = (t + a < 2048) ? memExt[t + a] : (_Float16)0.f;
  }

  v8f acc[8];
#pragma unroll
  for (int t = 0; t < 8; ++t) { v8f z = {}; acc[t] = z; }

  const int koffl = (lane < 16) ? 0 : 8;
  const int nloc  = lane & 15;

  // Per-bj column index and its (k-invariant) alignment class.
  int nn[4], acls[4];
#pragma unroll
  for (int bj = 0; bj < 4; ++bj) {
    nn[bj]   = n0 + joff + bj * 16 + nloc;
    acls[bj] = nn[bj] & 7;               // kb, koffl are multiples of 8
  }

  for (int kb = 0; kb < DIMD; kb += 32) {
    __syncthreads();                    // also covers initial rep visibility
    // Stage 128(i) x 32(k) block of K, f32 -> f16; k-pairs are contiguous in
    // global (mergeable b64 loads) and packed into one ds_store_b32.
#pragma unroll
    for (int it = 0; it < 8; ++it) {
      int e  = tid + it * 256;          // pair index: e = r*16 + kp
      int kp = e & 15, r = e >> 4;
      int k  = kp * 2;
      const float* src = &K[(long)(i0 + r) * DIMD + kb + k];
      *(unsigned*)&aT[r * LSTR + k] = pack2h(src[0], src[1]);
    }
    __syncthreads();

    v16h a0 = frag_ld(&aT[(ioff     ) * LSTR], lane);
    v16h a1 = frag_ld(&aT[(ioff + 16) * LSTR], lane);
#pragma unroll
    for (int bj = 0; bj < 4; ++bj) {
      const _Float16* rp = &rep[acls[bj] * RSTR];
      int basej = nn[bj] + kb + koffl - acls[bj];     // multiple of 8
      v8h lo = *(const v8h*)(rp + basej);             // memExt[nn+kbb+0..7]
      v8h hi = *(const v8h*)(rp + basej + 16);        // memExt[nn+kbb+16..23]
      v16h b = frag_cat(lo, hi);
      acc[0 * 4 + bj] = __builtin_amdgcn_wmma_f32_16x16x32_f16(
          false, a0, false, b, (short)0, acc[0 * 4 + bj], false, false);
      acc[1 * 4 + bj] = __builtin_amdgcn_wmma_f32_16x16x32_f16(
          false, a1, false, b, (short)0, acc[1 * 4 + bj], false, false);
    }
  }

  const int r16 = lane >> 4;
#pragma unroll
  for (int ai = 0; ai < 2; ++ai)
#pragma unroll
    for (int bj = 0; bj < 4; ++bj)
#pragma unroll
      for (int v = 0; v < 8; ++v) {
        long gi = i0 + ioff + ai * 16 + v + 8 * r16;
        int  gn = n0 + joff + bj * 16 + nloc;
        out[gi * DIMD + gn] = acc[ai * 4 + bj][v];
      }
}

// ---------------------------------------------------------------------------
extern "C" void kernel_launch(void* const* d_in, const int* in_sizes, int n_in,
                              void* d_out, int out_size, void* d_ws, size_t ws_size,
                              hipStream_t stream) {
  (void)in_sizes; (void)n_in; (void)out_size; (void)ws_size;
  const float* signal = (const float*)d_in[0];   // 32768 x 1024 f32
  const float* key    = (const float*)d_in[1];   // 32768 x 1024 f32
  const float* memory = (const float*)d_in[2];   // 1024 f32
  float* out = (float*)d_out;                    // 32768 x 1024 f32

  // Workspace layout (all re-written every launch; no cross-call state):
  //   [0, 32MB)      : 8 partial G buffers (split-K), f32
  //   [32MB, 36MB)   : reduced G, f32
  //   [36MB, +4KB)   : memExt, 2048 f16
  char* ws = (char*)d_ws;
  float*    Gp     = (float*)ws;
  float*    Gs     = (float*)(ws + (size_t)NSPLIT * DIMD * DIMD * sizeof(float));
  _Float16* memExt = (_Float16*)(ws + (size_t)(NSPLIT + 1) * DIMD * DIMD * sizeof(float));

  gemm1_corr <<<dim3(8, 8, NSPLIT),            256, 0, stream>>>(signal, key, Gp);
  reduceG    <<<dim3(DIMD * DIMD / 256, 1, 1), 256, 0, stream>>>(Gp, Gs);
  finalize_mem<<<dim3(1, 1, 1),               1024, 0, stream>>>(Gs, memory, memExt);
  gemm2_read <<<dim3(8, BATCH / 128, 1),       256, 0, stream>>>(key, memExt, out);
}
